// GraphSAGE_59356448031550
// MI455X (gfx1250) — compile-verified, matrix operand order checked by reading, not run on hardware
//
#include <hip/hip_runtime.h>
#include <math.h>

// ---------------------------------------------------------------------------
// GraphSAGE 2-layer, MI455X (gfx1250, wave32, WMMA).
//   h = relu(L2norm(mean_agg(x) @ Wl1^T + bl1 + x @ Wr1^T))
//   y =      L2norm(mean_agg(h) @ Wl2^T + bl2 + h @ Wr2^T)
// Scatter-mean via L2-resident fp32 atomics (agg = 51MB << 192MB L2).
// GEMMs + bias + L2-norm + relu fused into one bf16-WMMA kernel per layer.
// Weights are pre-converted ONCE to bf16 in the exact per-lane WMMA operand
// layout, so the hot kernel loads B tiles with zero VALU conversion work.
// ---------------------------------------------------------------------------

#define N_NODES 100000
#define N_EDGES 640000
#define D 128            // D_IN == D_HID == D_OUT == 128

typedef __attribute__((ext_vector_type(16))) __bf16 v16bf;
typedef __attribute__((ext_vector_type(8)))  float  v8f;

// Swizzled weight tile layout: tile (j, kt), lane l, element e  ->
//   flat[((j*4 + kt)*32 + l)*16 + e]
// so each lane's v16bf operand is one contiguous 32B load.
#define WTILE_ELEMS (8 * 4 * 32 * 16)   // 16384 bf16 per 128x128 matrix

// Map (lane, kt, e) -> K index per CDNA5 ISA 7.12.2 16-bit A/B layout:
// lanes 0-15: K = base..base+7 (e<8), base+16..base+23 (e>=8); lanes 16-31: +8.
__device__ __host__ inline int wmma_k_of(int lane, int kt, int e) {
  int base_k = kt * 32 + (lane >> 4) * 8;
  return (e < 8) ? (base_k + e) : (base_k + 16 + (e - 8));
}

// Load one lane's A slice (16x32 bf16 tile) from an fp32 row, fusing a scale.
// Both 8-element runs are 32B-contiguous -> global_load_b128 pairs.
__device__ inline v16bf load_A_bf16(const float* __restrict__ rowptr,
                                    int base_k, float scale) {
  v16bf a;
#pragma unroll
  for (int e = 0; e < 8; ++e)
    a[e] = (__bf16)(rowptr[base_k + e] * scale);
#pragma unroll
  for (int e = 0; e < 8; ++e)
    a[8 + e] = (__bf16)(rowptr[base_k + 16 + e] * scale);
  return a;
}

// ---------------------------------------------------------------------------
// One-time weight prep: fp32 [128,128] row-major -> bf16 swizzled tiles.
// B[k][n] = W[n][k]; tile columns n = j*16 + (lane&15).
// One thread per (tile, lane) slice of 16 elements.
// ---------------------------------------------------------------------------
__global__ void prep_weights(const float* __restrict__ W,
                             __bf16* __restrict__ Wt) {
  int t = blockIdx.x * blockDim.x + threadIdx.x;   // 0 .. 8*4*32-1
  if (t >= 8 * 4 * 32) return;
  int lane = t & 31;
  int kt   = (t >> 5) & 3;
  int j    = t >> 7;
  int n    = j * 16 + (lane & 15);
#pragma unroll
  for (int e = 0; e < 16; ++e) {
    int k = wmma_k_of(lane, kt, e);
    Wt[(size_t)t * 16 + e] = (__bf16)W[(size_t)n * D + k];
  }
}

// ---------------------------------------------------------------------------
// Scatter: one wave32 per edge; each lane moves float4 (128 feats / 32 lanes).
// agg[dst] += x[src]; lane0 optionally counts degree.
// ---------------------------------------------------------------------------
__global__ void sage_scatter(const float* __restrict__ x,
                             const int* __restrict__ src,
                             const int* __restrict__ dst,
                             float* __restrict__ agg,
                             float* __restrict__ cnt,
                             int addCount) {
  int tid  = blockIdx.x * blockDim.x + threadIdx.x;
  int e    = tid >> 5;
  int lane = tid & 31;
  if (e >= N_EDGES) return;
  int s = src[e];
  int d = dst[e];
  float4 v = ((const float4*)(x + (size_t)s * D))[lane];
  float* dp = agg + (size_t)d * D + lane * 4;
  atomicAdd(dp + 0, v.x);
  atomicAdd(dp + 1, v.y);
  atomicAdd(dp + 2, v.z);
  atomicAdd(dp + 3, v.w);
  if (addCount && lane == 0) atomicAdd(cnt + d, 1.0f);
}

// ---------------------------------------------------------------------------
// Fused SAGE layer: out = maybe_relu(L2norm(agg/max(cnt,1) @ Wl^T + bl + x @ Wr^T))
// One wave per 16 node rows; 8 column tiles x (4+4) v_wmma_f32_16x16x32_bf16.
// N_NODES % 16 == 0, so active waves always have full tiles (EXEC all-1s).
// B operands come pre-swizzled in bf16: one 32B contiguous load per tile.
// ---------------------------------------------------------------------------
__global__ void __launch_bounds__(256)
sage_wmma(const float* __restrict__ xin,
          const float* __restrict__ agg,
          const float* __restrict__ cnt,
          const v16bf* __restrict__ Wlt,  // swizzled bf16 tiles of Wl
          const float* __restrict__ bl,   // [D]
          const v16bf* __restrict__ Wrt,  // swizzled bf16 tiles of Wr
          float* __restrict__ out,        // [N, D]
          int relu) {
  const int wave    = threadIdx.x >> 5;
  const int rowbase = blockIdx.x * 128 + wave * 16;
  if (rowbase >= N_NODES) return;            // whole-wave uniform exit

  const int l    = threadIdx.x & 31;
  const int half = l >> 4;                   // K-group selector (A/B layout)
  const int lidx = l & 15;                   // A: row M; C/D: col N

  const float* aggrow = agg + (size_t)(rowbase + lidx) * D;
  const float* xrow   = xin + (size_t)(rowbase + lidx) * D;
  const float  inv    = 1.0f / fmaxf(cnt[rowbase + lidx], 1.0f);

  // Cache both A operands (4 K-tiles each) in registers; fuse 1/cnt scaling.
  v16bf Aagg[4], Ax[4];
#pragma unroll
  for (int kt = 0; kt < 4; ++kt) {
    int base_k = kt * 32 + half * 8;
    Aagg[kt] = load_A_bf16(aggrow, base_k, inv);
    Ax[kt]   = load_A_bf16(xrow,   base_k, 1.0f);
  }

  v8f   acc[8];
  float ss[8];
#pragma unroll
  for (int v = 0; v < 8; ++v) ss[v] = 0.0f;

#pragma unroll
  for (int j = 0; j < 8; ++j) {              // 8 column tiles of 16
    v8f c = {};
    // agg @ Wl^T : pre-swizzled bf16 B tiles, zero conversion VALU.
#pragma unroll
    for (int kt = 0; kt < 4; ++kt) {
      v16bf b = Wlt[(j * 4 + kt) * 32 + l];
      c = __builtin_amdgcn_wmma_f32_16x16x32_bf16(
              false, Aagg[kt], false, b, (short)0, c, false, false);
    }
    // + x @ Wr^T, chained into the same accumulator.
#pragma unroll
    for (int kt = 0; kt < 4; ++kt) {
      v16bf b = Wrt[(j * 4 + kt) * 32 + l];
      c = __builtin_amdgcn_wmma_f32_16x16x32_bf16(
              false, Ax[kt], false, b, (short)0, c, false, false);
    }
    // Bias (per output column N = lidx) + sum-of-squares partials.
    const float bias = bl[j * 16 + lidx];
#pragma unroll
    for (int v = 0; v < 8; ++v) {
      float t = c[v] + bias;
      c[v] = t;
      ss[v] += t * t;
    }
    acc[j] = c;
  }

  // Row sum-of-squares: reduce over the 16 lanes of each half (C/D layout:
  // row M = half*8 + v lives in lanes half*16 .. half*16+15). XOR masks 1..8
  // never cross the half boundary (bit 4).
#pragma unroll
  for (int mask = 1; mask < 16; mask <<= 1) {
#pragma unroll
    for (int v = 0; v < 8; ++v) ss[v] += __shfl_xor(ss[v], mask, 32);
  }
  float scale[8];
#pragma unroll
  for (int v = 0; v < 8; ++v)
    scale[v] = 1.0f / fmaxf(sqrtf(ss[v]), 1e-12f);

  // Epilogue: normalize (+ReLU) and store. Element (M, N) -> lane N + 16*(M>=8),
  // vgpr M&7.
#pragma unroll
  for (int j = 0; j < 8; ++j) {
#pragma unroll
    for (int v = 0; v < 8; ++v) {
      float t = acc[j][v] * scale[v];
      if (relu) t = fmaxf(t, 0.0f);
      int M = half * 8 + v;
      out[(size_t)(rowbase + M) * D + j * 16 + lidx] = t;
    }
  }
}

// ---------------------------------------------------------------------------
extern "C" void kernel_launch(void* const* d_in, const int* in_sizes, int n_in,
                              void* d_out, int out_size, void* d_ws, size_t ws_size,
                              hipStream_t stream) {
  const float* x    = (const float*)d_in[0];
  const int*   edge = (const int*)d_in[1];      // [2, E] flat
  const int*   src  = edge;
  const int*   dst  = edge + N_EDGES;
  const float* Wl1  = (const float*)d_in[2];
  const float* bl1  = (const float*)d_in[3];
  const float* Wr1  = (const float*)d_in[4];
  const float* Wl2  = (const float*)d_in[5];
  const float* bl2  = (const float*)d_in[6];
  const float* Wr2  = (const float*)d_in[7];
  float*       out  = (float*)d_out;

  // Workspace layout (52MB total):
  //   agg  [N,128] f32 | cnt [N] f32 | 4x swizzled bf16 weights (32KB each)
  float*  agg  = (float*)d_ws;
  float*  cnt  = agg + (size_t)N_NODES * D;
  __bf16* wb   = (__bf16*)(cnt + N_NODES);      // 32B-aligned (51.6e6 % 32 == 0)
  __bf16* Wl1t = wb;
  __bf16* Wr1t = wb + WTILE_ELEMS;
  __bf16* Wl2t = wb + 2 * WTILE_ELEMS;
  __bf16* Wr2t = wb + 3 * WTILE_ELEMS;

  const int scatterBlocks = (N_EDGES * 32) / 256;          // 80000
  const int sageBlocks    = (N_NODES + 127) / 128;         // 782
  const int prepBlocks    = (8 * 4 * 32 + 255) / 256;      // 4

  // One-time weight prep (tiny; overlaps nothing, costs ~nothing).
  prep_weights<<<prepBlocks, 256, 0, stream>>>(Wl1, Wl1t);
  prep_weights<<<prepBlocks, 256, 0, stream>>>(Wr1, Wr1t);
  prep_weights<<<prepBlocks, 256, 0, stream>>>(Wl2, Wl2t);
  prep_weights<<<prepBlocks, 256, 0, stream>>>(Wr2, Wr2t);

  // Layer 1: zero agg+cnt, scatter x (+degree), fused WMMA layer (ReLU).
  hipMemsetAsync(d_ws, 0, ((size_t)N_NODES * D + N_NODES) * sizeof(float), stream);
  sage_scatter<<<scatterBlocks, 256, 0, stream>>>(x, src, dst, agg, cnt, 1);
  sage_wmma<<<sageBlocks, 256, 0, stream>>>(x, agg, cnt, (const v16bf*)Wl1t, bl1,
                                            (const v16bf*)Wr1t, out, 1);

  // Layer 2: zero agg only (reuse degrees), scatter h (=out), fused layer.
  // In-place on out is safe: each wave reads only its own 16 rows of x before
  // writing those same rows.
  hipMemsetAsync(agg, 0, (size_t)N_NODES * D * sizeof(float), stream);
  sage_scatter<<<scatterBlocks, 256, 0, stream>>>(out, src, dst, agg, cnt, 0);
  sage_wmma<<<sageBlocks, 256, 0, stream>>>(out, agg, cnt, (const v16bf*)Wl2t, bl2,
                                            (const v16bf*)Wr2t, out, 0);
}
